// WindowAttention1D_16930761081043
// MI455X (gfx1250) — matrix operand order repaired and checked by reading, not compiled, not run on hardware
//
#include <hip/hip_runtime.h>

typedef __bf16 bf16;
typedef __bf16 v16bf __attribute__((ext_vector_type(16)));
typedef __bf16 v8bf  __attribute__((ext_vector_type(8)));
typedef __bf16 v4bf  __attribute__((ext_vector_type(4)));
typedef float  v8f   __attribute__((ext_vector_type(8)));
typedef float  v4f   __attribute__((ext_vector_type(4)));
typedef int    v4i   __attribute__((ext_vector_type(4)));

#define DIM   512
#define WIN   64
#define HEADS 16
#define HD    32

// ---- optional CDNA5 async global->LDS copy (ASYNCcnt path) ------------------
#if defined(__has_builtin)
#if __has_builtin(__builtin_amdgcn_global_load_async_to_lds_b128)
#define USE_ASYNC_LDS 1
#endif
#endif

#ifdef USE_ASYNC_LDS
typedef __attribute__((address_space(1))) v4i gv4i;   // global int4
typedef __attribute__((address_space(3))) v4i lv4i;   // LDS int4
__device__ __forceinline__ void async_cp16(const void* g, void* l) {
  // copies 16B per active lane, memory -> LDS, tracked by ASYNCcnt
  __builtin_amdgcn_global_load_async_to_lds_b128((gv4i*)g, (lv4i*)l, 0, 0);
}
__device__ __forceinline__ void async_wait0() {
#if __has_builtin(__builtin_amdgcn_s_wait_asynccnt)
  __builtin_amdgcn_s_wait_asynccnt(0);
#else
  asm volatile("s_wait_asynccnt 0x0" ::: "memory");
#endif
}
#endif

// ---------------- WMMA fragment helpers (CDNA5 wave32 layouts) ----------------
// A-matrix 16x32 bf16: lane = row (lane&15), half = lane>>4.
//   elements 0..7  -> K = half*8 + 0..7      (contiguous 16B)
//   elements 8..15 -> K = 16 + half*8 + 0..7 (contiguous 16B)
__device__ __forceinline__ v16bf frag_a(const bf16* rowk, int half) {
  v8bf lo = *(const v8bf*)(rowk + half * 8);
  v8bf hi = *(const v8bf*)(rowk + 16 + half * 8);
  v16bf a;
#pragma unroll
  for (int i = 0; i < 8; ++i) { a[i] = lo[i]; a[i + 8] = hi[i]; }
  return a;
}

// B-matrix 32x16 bf16: lane = col (lane&15), half = lane>>4.
//   elements 0..7  -> K = half*8 + 0..7, elements 8..15 -> K = 16 + half*8 + 0..7
__device__ __forceinline__ v16bf frag_b(const bf16* colk, int half) {
  v8bf lo = *(const v8bf*)(colk + half * 8);
  v8bf hi = *(const v8bf*)(colk + 16 + half * 8);
  v16bf b;
#pragma unroll
  for (int i = 0; i < 8; ++i) { b[i] = lo[i]; b[i + 8] = hi[i]; }
  return b;
}

__device__ __forceinline__ v8f zero8() {
  v8f z;
#pragma unroll
  for (int i = 0; i < 8; ++i) z[i] = 0.0f;
  return z;
}

__device__ __forceinline__ v8f wmma_bf16(v16bf a, v16bf b, v8f c) {
  // 8 args: (neg_a, A, neg_b, B, c_mod, C, reuse_a, reuse_b)
  return __builtin_amdgcn_wmma_f32_16x16x32_bf16(false, a, false, b, (short)0, c,
                                                 false, false);
}

// ---------------- kernel 0: fp32 -> bf16 weight conversion (L2-resident) -----
__global__ __launch_bounds__(256) void convert_weights_kernel(
    const float* __restrict__ wq, const float* __restrict__ wp,
    bf16* __restrict__ oq, bf16* __restrict__ op_) {
  int i = blockIdx.x * blockDim.x + threadIdx.x;
  if (i < 3 * DIM * DIM) oq[i] = (bf16)wq[i];
  if (i < DIM * DIM)     op_[i] = (bf16)wp[i];
}

// ---------------- fused window-attention kernel: one block per window --------
// LDS layout (bytes):
//   [  0K.. 64K) sQ [16][64][32] bf16   attn-out overwrites after PV
//   [ 64K..128K) sK [16][64][32] bf16
//   [128K..192K) sV [16][32][64] bf16   (transposed: PV B-frags k-contiguous)
//   [192K..256K) sX [64][512]   bf16    x window; reused as P [8 waves][64][64]
//   [256K..272K) sMask [64][64] f32     shift mask window
//   [272K..280K) sRel  [127][16] f32    relative-position bias table
__global__ __launch_bounds__(256) void fused_window_attn_kernel(
    const float* __restrict__ x,     const float* __restrict__ mask,
    const bf16*  __restrict__ wqkv,  const float* __restrict__ bqkv,
    const bf16*  __restrict__ wproj, const float* __restrict__ bproj,
    const float* __restrict__ rel,   float* __restrict__ out) {
  extern __shared__ __align__(16) char smem_raw[];
  bf16* sQ = (bf16*)smem_raw;                 // 32768 elems
  bf16* sK = sQ + HEADS * WIN * HD;           // 32768
  bf16* sV = sK + HEADS * WIN * HD;           // 32768
  bf16* sX = sV + HEADS * WIN * HD;           // 32768
  float* sMask = (float*)(smem_raw + 262144); // 4096 f32
  float* sRel  = sMask + WIN * WIN;           // 2032 f32 (127*16)

  const int b    = blockIdx.x;
  const int tid  = threadIdx.x;
  // wave id is wave-uniform: pin it to an SGPR so derived values go scalar
  const int w    = __builtin_amdgcn_readfirstlane(tid >> 5);   // 0..7
  const int lane = tid & 31;
  const int lc   = lane & 15;
  const int half = lane >> 4;

  // ---- stage mask + rel via async global->LDS (raw copy, ASYNCcnt) ---------
  const float* msrc = mask + (size_t)(b & 63) * (WIN * WIN);
#ifdef USE_ASYNC_LDS
  {
#pragma unroll
    for (int i = 0; i < 4; ++i) {
      const int c = tid + i * 256;          // 16B chunk id, 1024 total
      async_cp16(msrc + c * 4, sMask + c * 4);
    }
#pragma unroll
    for (int i = 0; i < 2; ++i) {
      const int c = tid + i * 256;          // 508 x 16B chunks (2032 f32)
      if (c < ((2 * WIN - 1) * HEADS) / 4) async_cp16(rel + c * 4, sRel + c * 4);
    }
  }
#else
  {
#pragma unroll
    for (int i = 0; i < 4; ++i)
      *(v4f*)(sMask + tid * 16 + i * 4) = *(const v4f*)(msrc + tid * 16 + i * 4);
#pragma unroll
    for (int i = 0; i < 8; ++i) {
      const int idx = tid * 8 + i;
      if (idx < (2 * WIN - 1) * HEADS) sRel[idx] = rel[idx];
    }
  }
#endif

  // ---- stage x window (fp32 -> bf16 conversion in flight) -------------------
  {
    const int row = tid >> 2;
    const int c0  = (tid & 3) * 128;
    const float* src = x + ((size_t)b * WIN + row) * DIM + c0;
    bf16* dst = sX + row * DIM + c0;
#pragma unroll 8
    for (int i = 0; i < 32; ++i) {
      v4f f = *(const v4f*)(src + i * 4);
      v4bf h;
#pragma unroll
      for (int e = 0; e < 4; ++e) h[e] = (bf16)f[e];
      *(v4bf*)(dst + i * 4) = h;
    }
  }
#ifdef USE_ASYNC_LDS
  async_wait0();   // retire async LDS writes before the barrier
#endif
  __syncthreads();

  // ---- QKV GEMM: 64x1536 = x(64x512) @ Wqkv^T, 6 passes of 256 cols ---------
  const float scale = 0.17677669529663687f;  // 1/sqrt(32)
#pragma unroll 1
  for (int p = 0; p < 6; ++p) {
    const int c0     = __builtin_amdgcn_readfirstlane(p * 256 + w * 32);
    const int region = __builtin_amdgcn_readfirstlane(c0 / DIM);   // 0=Q 1=K 2=V
    const int h      = __builtin_amdgcn_readfirstlane((c0 % DIM) / HD);
    v8f acc[4][2];
#pragma unroll
    for (int mt = 0; mt < 4; ++mt)
#pragma unroll
      for (int nt = 0; nt < 2; ++nt) acc[mt][nt] = zero8();

#pragma unroll 1
    for (int kk = 0; kk < DIM; kk += 32) {
      v16bf b0 = frag_b(wqkv + (size_t)(c0 + lc) * DIM + kk, half);
      v16bf b1 = frag_b(wqkv + (size_t)(c0 + 16 + lc) * DIM + kk, half);
#pragma unroll
      for (int mt = 0; mt < 4; ++mt) {
        v16bf a = frag_a(sX + (mt * 16 + lc) * DIM + kk, half);
        acc[mt][0] = wmma_bf16(a, b0, acc[mt][0]);
        acc[mt][1] = wmma_bf16(a, b1, acc[mt][1]);
      }
    }

    // epilogue: +bias, (scale if Q), scatter into per-head LDS layouts.
    // region is an SGPR -> scalar branch; each path has compile-time strides so
    // stores are ds_store_b16 with immediate offsets off one base per nt.
    if (region == 0) {
      bf16* dstq = sQ + h * (WIN * HD) + half * 8 * HD;
#pragma unroll
      for (int nt = 0; nt < 2; ++nt) {
        const float bias = bqkv[c0 + nt * 16 + lc];
        bf16* pbase = dstq + nt * 16 + lc;           // + hd
#pragma unroll
        for (int mt = 0; mt < 4; ++mt)
#pragma unroll
          for (int r = 0; r < 8; ++r)
            pbase[(mt * 16 + r) * HD] = (bf16)((acc[mt][nt][r] + bias) * scale);
      }
    } else if (region == 1) {
      bf16* dstk = sK + h * (WIN * HD) + half * 8 * HD;
#pragma unroll
      for (int nt = 0; nt < 2; ++nt) {
        const float bias = bqkv[c0 + nt * 16 + lc];
        bf16* pbase = dstk + nt * 16 + lc;
#pragma unroll
        for (int mt = 0; mt < 4; ++mt)
#pragma unroll
          for (int r = 0; r < 8; ++r)
            pbase[(mt * 16 + r) * HD] = (bf16)(acc[mt][nt][r] + bias);
      }
    } else {
      // V transposed: [hd][row]
      bf16* dstv = sV + h * (WIN * HD) + half * 8;
#pragma unroll
      for (int nt = 0; nt < 2; ++nt) {
        const float bias = bqkv[c0 + nt * 16 + lc];
        bf16* pbase = dstv + (nt * 16 + lc) * WIN;   // + hd*WIN
#pragma unroll
        for (int mt = 0; mt < 4; ++mt)
#pragma unroll
          for (int r = 0; r < 8; ++r)
            pbase[mt * 16 + r] = (bf16)(acc[mt][nt][r] + bias);
      }
    }
  }
  __syncthreads();   // everyone done with sX before P reuse

  // ---- attention: wave w handles heads {w, w+8} -----------------------------
  bf16* sP = sX + w * (WIN * WIN);           // this wave's 64x64 P tile

#pragma unroll 1
  for (int hh = 0; hh < 2; ++hh) {
    const int h = __builtin_amdgcn_readfirstlane(w + hh * 8);
    const bf16* qh = sQ + h * (WIN * HD);
    const bf16* kh = sK + h * (WIN * HD);
    const bf16* vh = sV + h * (WIN * HD);

    // S = Q @ K^T : 64x64, single K-step (hd = 32)
    v16bf qa[4], kb[4];
#pragma unroll
    for (int mt = 0; mt < 4; ++mt) qa[mt] = frag_a(qh + (mt * 16 + lc) * HD, half);
#pragma unroll
    for (int nt = 0; nt < 4; ++nt) kb[nt] = frag_b(kh + (nt * 16 + lc) * HD, half);
    v8f s[4][4];
#pragma unroll
    for (int mt = 0; mt < 4; ++mt)
#pragma unroll
      for (int nt = 0; nt < 4; ++nt)
        s[mt][nt] = wmma_bf16(qa[mt], kb[nt], zero8());

    // + relative-position bias + shift mask (both LDS-resident), row softmax
#pragma unroll
    for (int mt = 0; mt < 4; ++mt) {
#pragma unroll
      for (int nt = 0; nt < 4; ++nt)
#pragma unroll
        for (int r = 0; r < 8; ++r) {
          const int i = mt * 16 + r + half * 8;
          const int j = nt * 16 + lc;
          s[mt][nt][r] += sRel[(i - j + WIN - 1) * HEADS + h] + sMask[i * WIN + j];
        }
#pragma unroll
      for (int r = 0; r < 8; ++r) {
        float m = fmaxf(fmaxf(s[mt][0][r], s[mt][1][r]),
                        fmaxf(s[mt][2][r], s[mt][3][r]));
        m = fmaxf(m, __shfl_xor(m, 1, 32));
        m = fmaxf(m, __shfl_xor(m, 2, 32));
        m = fmaxf(m, __shfl_xor(m, 4, 32));
        m = fmaxf(m, __shfl_xor(m, 8, 32));
        float sum = 0.0f;
#pragma unroll
        for (int nt = 0; nt < 4; ++nt) {
          float e = __expf(s[mt][nt][r] - m);
          s[mt][nt][r] = e;
          sum += e;
        }
        sum += __shfl_xor(sum, 1, 32);
        sum += __shfl_xor(sum, 2, 32);
        sum += __shfl_xor(sum, 4, 32);
        sum += __shfl_xor(sum, 8, 32);
        const float inv = 1.0f / sum;
#pragma unroll
        for (int nt = 0; nt < 4; ++nt) s[mt][nt][r] *= inv;
      }
      // write P rows for this row-block (wave-local LDS, HW keeps DS in order)
      {
        bf16* pbase = sP + (mt * 16 + half * 8) * WIN + lc;
#pragma unroll
        for (int nt = 0; nt < 4; ++nt)
#pragma unroll
          for (int r = 0; r < 8; ++r)
            pbase[r * WIN + nt * 16] = (bf16)s[mt][nt][r];
      }
    }

    // O = P @ V : 64x32, K = 64 in two 32-steps
    v8f o[4][2];
#pragma unroll
    for (int mt = 0; mt < 4; ++mt) { o[mt][0] = zero8(); o[mt][1] = zero8(); }
#pragma unroll
    for (int kt = 0; kt < 2; ++kt) {
      v16bf vb0 = frag_b(vh + (lc) * WIN + kt * 32, half);
      v16bf vb1 = frag_b(vh + (16 + lc) * WIN + kt * 32, half);
#pragma unroll
      for (int mt = 0; mt < 4; ++mt) {
        v16bf pa = frag_a(sP + (mt * 16 + lc) * WIN + kt * 32, half);
        o[mt][0] = wmma_bf16(pa, vb0, o[mt][0]);
        o[mt][1] = wmma_bf16(pa, vb1, o[mt][1]);
      }
    }
    // store attn-out over the (now dead) Q region: layout [row][hd] per head
    {
      bf16* abase = sQ + h * (WIN * HD) + half * 8 * HD;
#pragma unroll
      for (int nt = 0; nt < 2; ++nt) {
        bf16* pbase = abase + nt * 16 + lc;
#pragma unroll
        for (int mt = 0; mt < 4; ++mt)
#pragma unroll
          for (int r = 0; r < 8; ++r)
            pbase[(mt * 16 + r) * HD] = (bf16)o[mt][nt][r];
      }
    }
  }
  __syncthreads();   // all heads' attn-out visible before proj

  // ---- proj GEMM: out(64x512) = attnout(64x512) @ Wproj^T + b ---------------
  {
    const int c0 = __builtin_amdgcn_readfirstlane(w * 64);
    v8f acc[4][4];
#pragma unroll
    for (int mt = 0; mt < 4; ++mt)
#pragma unroll
      for (int nt = 0; nt < 4; ++nt) acc[mt][nt] = zero8();

#pragma unroll 1
    for (int kk = 0; kk < DIM; kk += 32) {
      const bf16* abase = sQ + (kk >> 5) * (WIN * HD);  // head-aligned K block
      v16bf bfr[4];
#pragma unroll
      for (int nt = 0; nt < 4; ++nt)
        bfr[nt] = frag_b(wproj + (size_t)(c0 + nt * 16 + lc) * DIM + kk, half);
#pragma unroll
      for (int mt = 0; mt < 4; ++mt) {
        v16bf a = frag_a(abase + (mt * 16 + lc) * HD, half);
#pragma unroll
        for (int nt = 0; nt < 4; ++nt)
          acc[mt][nt] = wmma_bf16(a, bfr[nt], acc[mt][nt]);
      }
    }

#pragma unroll
    for (int nt = 0; nt < 4; ++nt) {
      const int col = c0 + nt * 16 + lc;
      const float bias = bproj[col];
      float* obase = out + ((size_t)b * WIN + half * 8) * DIM + col;
#pragma unroll
      for (int mt = 0; mt < 4; ++mt)
#pragma unroll
        for (int r = 0; r < 8; ++r)
          obase[(size_t)(mt * 16 + r) * DIM] = acc[mt][nt][r] + bias;
    }
  }
}

// ---------------- host launcher ----------------------------------------------
extern "C" void kernel_launch(void* const* d_in, const int* in_sizes, int n_in,
                              void* d_out, int out_size, void* d_ws, size_t ws_size,
                              hipStream_t stream) {
  (void)in_sizes; (void)n_in; (void)out_size; (void)ws_size;
  const float* x      = (const float*)d_in[0];
  const float* mask   = (const float*)d_in[1];
  const float* qkv_w  = (const float*)d_in[2];
  const float* qkv_b  = (const float*)d_in[3];
  const float* proj_w = (const float*)d_in[4];
  const float* proj_b = (const float*)d_in[5];
  const float* rel    = (const float*)d_in[6];
  float* out = (float*)d_out;

  bf16* wq_bf = (bf16*)d_ws;                    // 3*512*512 bf16
  bf16* wp_bf = wq_bf + 3 * DIM * DIM;          // 512*512 bf16

  convert_weights_kernel<<<(3 * DIM * DIM + 255) / 256, 256, 0, stream>>>(
      qkv_w, proj_w, wq_bf, wp_bf);

  // 256KB bf16 tiles + 16KB mask + 8KB rel = 280KB dynamic LDS (<=320KB/WGP)
  const size_t smem = 4 * HEADS * WIN * HD * sizeof(bf16)
                    + (WIN * WIN + (2 * WIN - 1) * HEADS) * sizeof(float);
  (void)hipFuncSetAttribute((const void*)fused_window_attn_kernel,
                            hipFuncAttributeMaxDynamicSharedMemorySize,
                            (int)smem);
  fused_window_attn_kernel<<<4096, 256, smem, stream>>>(
      x, mask, wq_bf, qkv_b, wp_bf, proj_b, rel, out);
}
